// TransformerDecoderLayer_65403761983682
// MI455X (gfx1250) — matrix operand order repaired and checked by reading, not compiled, not run on hardware
//
#include <hip/hip_runtime.h>

typedef _Float16 half_t;
typedef __attribute__((ext_vector_type(4)))  _Float16 v4h;
typedef __attribute__((ext_vector_type(8)))  _Float16 v8h;
typedef __attribute__((ext_vector_type(16))) _Float16 v16h;
typedef __attribute__((ext_vector_type(8)))  float    v8f;

#define D_MODEL 256
#define N_HEADS 8
#define HDIM    32
#define N_LVL   3
#define N_PTS   4
#define BSZ     32
#define LQN     300
#define LVN     8400
#define DFFN    1024
#define NBQ     (BSZ * LQN)      /* 9600 token rows   */
#define NMV     (BSZ * LVN)      /* 268800 value rows */

// ---------------------------------------------------------------------------
// Tiled WMMA GEMM:  C[M,N] = epi( A(+A2)[M,K] @ W[K,N] + bias[N] )
// Block tile 128x64, 8 waves; each wave computes 16x64 with 4 accumulators
// (one shared A fragment -> 4 x v_wmma_f32_16x16x32_f16 per K-step).
// Double-buffered LDS software pipeline: next tile's global_load_b128s are
// issued before the current tile's WMMAs, converted f32->f16 and stored into
// the alternate buffer afterwards; one barrier per K-step.
// LDS layouts are fragment-native so every fragment is 2 x ds_load_b128:
//   As[buf][row][k]  row stride 40 halves (80 B, 16B-aligned)
//   Bt[buf][n][k]    transposed W tile, same stride
// Requires: M % 128 == 0, K % 32 == 0, N % 4 == 0 (true for all calls).
// ---------------------------------------------------------------------------
template<bool FUSE, bool RELU, bool OUT_F16>
__global__ __launch_bounds__(256)
void gemm_wmma(const float* __restrict__ A, const float* __restrict__ A2,
               const float* __restrict__ W, const float* __restrict__ bias,
               void* __restrict__ Cout, int M, int N, int K)
{
    __shared__ __align__(16) half_t As[2][128][40];
    __shared__ __align__(16) half_t Bt[2][64][40];

    const int tid  = threadIdx.x;
    const int wave = tid >> 5;         // 0..7 : 16-row slice of the 128-row tile
    const int lane = tid & 31;
    const int m0   = blockIdx.x * 128;
    const int n0   = blockIdx.y * 64;

    const int half_sel = (lane >> 4) & 1;   // lane-half selects K subgroup
    const int arow     = wave * 16 + (lane & 15);
    const int ncol     = lane & 15;

    // cooperative-load coordinates
    const int ar = tid >> 3;           // 0..31 (A rows, 4 passes of 32)
    const int ac = (tid & 7) * 4;      // A col group (float4)
    const int br = tid >> 4;           // 0..15 (W rows, 2 passes of 16)
    const int bc = (tid & 15) * 4;     // W col group (float4)
    const bool bvalid = (n0 + bc) < N; // N % 4 == 0 -> whole vector in/out

    v8f acc0 = {}, acc1 = {}, acc2 = {}, acc3 = {};

    float4 ra[4];                      // staged A tile (post-fuse, f32)
    float4 rb[2];                      // staged W tile (f32)

    auto load_tile = [&](int k0) {
        #pragma unroll
        for (int p = 0; p < 4; ++p) {
            const size_t gidx = (size_t)(m0 + ar + p * 32) * K + (k0 + ac);
            float4 v = *(const float4*)(A + gidx);
            if (FUSE) {
                const float4 v2 = *(const float4*)(A2 + gidx);
                v.x += v2.x; v.y += v2.y; v.z += v2.z; v.w += v2.w;
            }
            ra[p] = v;
        }
        #pragma unroll
        for (int p = 0; p < 2; ++p) {
            float4 v = make_float4(0.f, 0.f, 0.f, 0.f);
            if (bvalid)
                v = *(const float4*)(W + (size_t)(k0 + br + p * 16) * N + (n0 + bc));
            rb[p] = v;
        }
    };

    auto store_tile = [&](int buf) {
        #pragma unroll
        for (int p = 0; p < 4; ++p) {
            v4h h;
            h[0] = (half_t)ra[p].x; h[1] = (half_t)ra[p].y;
            h[2] = (half_t)ra[p].z; h[3] = (half_t)ra[p].w;
            *(v4h*)&As[buf][ar + p * 32][ac] = h;
        }
        #pragma unroll
        for (int p = 0; p < 2; ++p) {
            const int r = br + p * 16;
            Bt[buf][bc + 0][r] = (half_t)rb[p].x;
            Bt[buf][bc + 1][r] = (half_t)rb[p].y;
            Bt[buf][bc + 2][r] = (half_t)rb[p].z;
            Bt[buf][bc + 3][r] = (half_t)rb[p].w;
        }
    };

    auto compute = [&](int buf) {
        const v8h* apv = (const v8h*)&As[buf][arow][0];
        const v8h  alo = apv[half_sel * 2];      // halves 16*half_sel .. +7
        const v8h  ahi = apv[half_sel * 2 + 1];  // halves 16*half_sel+8 .. +15
        const v16h a   = __builtin_shufflevector(alo, ahi,
                           0,1,2,3,4,5,6,7,8,9,10,11,12,13,14,15);
        const int  ksel = half_sel * 2;          // halves 16*half_sel .. +15

        const v8h* b0p = (const v8h*)&Bt[buf][ 0 + ncol][0];
        const v16h b0 = __builtin_shufflevector(b0p[ksel], b0p[ksel + 1],
                           0,1,2,3,4,5,6,7,8,9,10,11,12,13,14,15);
        acc0 = __builtin_amdgcn_wmma_f32_16x16x32_f16(false, a, false, b0,
                                                      (short)0, acc0, false, false);
        const v8h* b1p = (const v8h*)&Bt[buf][16 + ncol][0];
        const v16h b1 = __builtin_shufflevector(b1p[ksel], b1p[ksel + 1],
                           0,1,2,3,4,5,6,7,8,9,10,11,12,13,14,15);
        acc1 = __builtin_amdgcn_wmma_f32_16x16x32_f16(false, a, false, b1,
                                                      (short)0, acc1, false, false);
        const v8h* b2p = (const v8h*)&Bt[buf][32 + ncol][0];
        const v16h b2 = __builtin_shufflevector(b2p[ksel], b2p[ksel + 1],
                           0,1,2,3,4,5,6,7,8,9,10,11,12,13,14,15);
        acc2 = __builtin_amdgcn_wmma_f32_16x16x32_f16(false, a, false, b2,
                                                      (short)0, acc2, false, false);
        const v8h* b3p = (const v8h*)&Bt[buf][48 + ncol][0];
        const v16h b3 = __builtin_shufflevector(b3p[ksel], b3p[ksel + 1],
                           0,1,2,3,4,5,6,7,8,9,10,11,12,13,14,15);
        acc3 = __builtin_amdgcn_wmma_f32_16x16x32_f16(false, a, false, b3,
                                                      (short)0, acc3, false, false);
    };

    // ---- software pipeline ----
    load_tile(0);
    store_tile(0);
    __syncthreads();

    const int nk = K >> 5;
    for (int i = 0; i < nk; ++i) {
        if (i + 1 < nk) load_tile((i + 1) << 5);   // globals in flight...
        compute(i & 1);                            // ...while WMMAs issue
        if (i + 1 < nk) store_tile((i + 1) & 1);   // fill the other buffer
        __syncthreads();
    }

    // ---- epilogue: lane holds col (lane&15), rows r + 8*half_sel ----
    const int rbase = half_sel * 8;
    const v8f accs[4] = {acc0, acc1, acc2, acc3};
    #pragma unroll
    for (int t = 0; t < 4; ++t) {
        const int n = n0 + t * 16 + ncol;
        if (n >= N) continue;
        const float bb = bias[n];
        #pragma unroll
        for (int r = 0; r < 8; ++r) {
            const int gm = m0 + wave * 16 + rbase + r;
            float o = accs[t][r] + bb;
            if (RELU) o = fmaxf(o, 0.0f);
            if (OUT_F16) ((half_t*)Cout)[(size_t)gm * N + n] = (half_t)o;
            else         ((float*)Cout)[(size_t)gm * N + n] = o;
        }
    }
}

// ---------------------------------------------------------------------------
// Self-attention: one block per (b, h).  K/V staged in LDS (f16),
// per-thread online softmax over the 300 keys, hd=32 accumulator.
// ---------------------------------------------------------------------------
__global__ __launch_bounds__(256)
void attn_self(const float* __restrict__ Q, const float* __restrict__ Kb,
               const float* __restrict__ Vb, float* __restrict__ O)
{
    const int b = blockIdx.x >> 3;
    const int h = blockIdx.x & 7;

    __shared__ half_t Ks[LQN * HDIM];
    __shared__ half_t Vs[LQN * HDIM];

    for (int i = threadIdx.x; i < LQN * HDIM; i += 256) {
        int r = i >> 5, c = i & 31;
        size_t g = ((size_t)(b * LQN + r)) * D_MODEL + h * HDIM + c;
        Ks[i] = (half_t)Kb[g];
        Vs[i] = (half_t)Vb[g];
    }
    __syncthreads();

    const float scale = 0.17677669529663687f; // 1/sqrt(32)
    for (int q = threadIdx.x; q < LQN; q += 256) {
        float qv[HDIM];
        size_t qbase = ((size_t)(b * LQN + q)) * D_MODEL + h * HDIM;
        #pragma unroll
        for (int c = 0; c < HDIM; ++c) qv[c] = Q[qbase + c] * scale;

        float m = -1e30f, s = 0.0f;
        float acc[HDIM];
        #pragma unroll
        for (int c = 0; c < HDIM; ++c) acc[c] = 0.0f;

        for (int k = 0; k < LQN; ++k) {
            float d = 0.0f;
            #pragma unroll
            for (int c = 0; c < HDIM; ++c) d += qv[c] * (float)Ks[k * HDIM + c];
            float mn   = fmaxf(m, d);
            float corr = __expf(m - mn);
            float p    = __expf(d - mn);
            s = s * corr + p;
            #pragma unroll
            for (int c = 0; c < HDIM; ++c)
                acc[c] = acc[c] * corr + p * (float)Vs[k * HDIM + c];
            m = mn;
        }
        float inv = 1.0f / s;
        #pragma unroll
        for (int c = 0; c < HDIM; ++c) O[qbase + c] = acc[c] * inv;
    }
}

// ---------------------------------------------------------------------------
// LayerNorm(x + res) * g + b   — one wave (32 lanes) per 256-wide row.
// ---------------------------------------------------------------------------
__global__ __launch_bounds__(256)
void ln_res(const float* __restrict__ x, const float* __restrict__ res,
            const float* __restrict__ g, const float* __restrict__ be,
            float* __restrict__ y, int rows)
{
    const int wave = threadIdx.x >> 5;
    const int lane = threadIdx.x & 31;
    const int row  = blockIdx.x * 8 + wave;
    if (row >= rows) return;

    const size_t base = (size_t)row * D_MODEL;
    float v[8];
    float sum = 0.0f;
    #pragma unroll
    for (int i = 0; i < 8; ++i) {
        int c = lane + 32 * i;
        float t = x[base + c];
        if (res) t += res[base + c];
        v[i] = t; sum += t;
    }
    #pragma unroll
    for (int o = 16; o > 0; o >>= 1) sum += __shfl_xor(sum, o, 32);
    float mean = sum * (1.0f / 256.0f);

    float var = 0.0f;
    #pragma unroll
    for (int i = 0; i < 8; ++i) { float d = v[i] - mean; var += d * d; }
    #pragma unroll
    for (int o = 16; o > 0; o >>= 1) var += __shfl_xor(var, o, 32);
    var *= (1.0f / 256.0f);
    float rstd = rsqrtf(var + 1e-5f);

    #pragma unroll
    for (int i = 0; i < 8; ++i) {
        int c = lane + 32 * i;
        y[base + c] = (v[i] - mean) * rstd * g[c] + be[c];
    }
}

// ---------------------------------------------------------------------------
// MS-deformable sampling.  Block = one (b,q); wave = head; lane = channel.
// Softmax over the 12 (level,point) logits in-thread, bilinear gather from
// the f16 value tensor.
// ---------------------------------------------------------------------------
__global__ __launch_bounds__(256)
void deform_sample(const half_t* __restrict__ value, const float* __restrict__ refp,
                   const float* __restrict__ off, const float* __restrict__ logit,
                   float* __restrict__ out)
{
    const int bq = blockIdx.x;           // 0..NBQ-1
    const int b  = bq / LQN;
    const int h  = threadIdx.x >> 5;
    const int c  = threadIdx.x & 31;

    const float* ob = off   + (size_t)bq * (N_HEADS * N_LVL * N_PTS * 2) + h * (N_LVL * N_PTS * 2);
    const float* lb = logit + (size_t)bq * (N_HEADS * N_LVL * N_PTS)     + h * (N_LVL * N_PTS);
    const float* rp = refp  + (size_t)bq * (N_LVL * 2);

    // softmax over 12 logits
    float e[12];
    float lmax = -1e30f;
    #pragma unroll
    for (int i = 0; i < 12; ++i) lmax = fmaxf(lmax, lb[i]);
    float s = 0.0f;
    #pragma unroll
    for (int i = 0; i < 12; ++i) { e[i] = __expf(lb[i] - lmax); s += e[i]; }
    const float inv = 1.0f / s;

    const int Hs[N_LVL] = {80, 40, 20};
    const int Ws[N_LVL] = {80, 40, 20};
    const int St[N_LVL] = {0, 6400, 8000};

    float acc = 0.0f;
    #pragma unroll
    for (int l = 0; l < N_LVL; ++l) {
        const float wl = (float)Ws[l], hl = (float)Hs[l];
        const float rx = rp[l * 2 + 0], ry = rp[l * 2 + 1];
        const int   Wi = Ws[l], Hi = Hs[l], S0 = St[l];
        #pragma unroll
        for (int p = 0; p < N_PTS; ++p) {
            const float aw = e[l * N_PTS + p] * inv;
            const float lx = rx + ob[(l * N_PTS + p) * 2 + 0] / wl;
            const float ly = ry + ob[(l * N_PTS + p) * 2 + 1] / hl;
            const float x  = lx * wl - 0.5f;
            const float y  = ly * hl - 0.5f;
            const float xf = floorf(x), yf = floorf(y);
            const int   x0 = (int)xf,  y0 = (int)yf;
            const float wx = x - xf,   wy = y - yf;

            const float cw[4] = { aw * (1.0f - wx) * (1.0f - wy),
                                  aw * wx * (1.0f - wy),
                                  aw * (1.0f - wx) * wy,
                                  aw * wx * wy };
            const int cx[4] = { x0, x0 + 1, x0,     x0 + 1 };
            const int cy[4] = { y0, y0,     y0 + 1, y0 + 1 };
            #pragma unroll
            for (int t = 0; t < 4; ++t) {
                int xi = cx[t], yi = cy[t];
                if (xi >= 0 && xi < Wi && yi >= 0 && yi < Hi) {
                    size_t idx = ((size_t)(b * LVN + S0 + yi * Wi + xi)) * D_MODEL
                               + h * HDIM + c;
                    acc += cw[t] * (float)value[idx];
                }
            }
        }
    }
    out[(size_t)bq * D_MODEL + h * HDIM + c] = acc;
}

// ---------------------------------------------------------------------------
extern "C" void kernel_launch(void* const* d_in, const int* in_sizes, int n_in,
                              void* d_out, int out_size, void* d_ws, size_t ws_size,
                              hipStream_t stream)
{
    (void)in_sizes; (void)n_in; (void)out_size; (void)ws_size;

    const float* tgt   = (const float*)d_in[0];
    const float* refp  = (const float*)d_in[1];
    const float* mem   = (const float*)d_in[2];
    const float* qpe   = (const float*)d_in[3];
    // d_in[4], d_in[5]: spatial shapes / level starts (compile-time constants)
    const float* Wq    = (const float*)d_in[6];  const float* bq_   = (const float*)d_in[7];
    const float* Wk    = (const float*)d_in[8];  const float* bk_   = (const float*)d_in[9];
    const float* Wv    = (const float*)d_in[10]; const float* bv_   = (const float*)d_in[11];
    const float* Wo    = (const float*)d_in[12]; const float* bo_   = (const float*)d_in[13];
    const float* ln1g  = (const float*)d_in[14]; const float* ln1b  = (const float*)d_in[15];
    const float* Wvp   = (const float*)d_in[16]; const float* bvp   = (const float*)d_in[17];
    const float* Woff  = (const float*)d_in[18]; const float* boff  = (const float*)d_in[19];
    const float* Wattn = (const float*)d_in[20]; const float* battn = (const float*)d_in[21];
    const float* Wout  = (const float*)d_in[22]; const float* bout  = (const float*)d_in[23];
    const float* ln2g  = (const float*)d_in[24]; const float* ln2b  = (const float*)d_in[25];
    const float* W1    = (const float*)d_in[26]; const float* b1_   = (const float*)d_in[27];
    const float* W2    = (const float*)d_in[28]; const float* b2_   = (const float*)d_in[29];
    const float* ln3g  = (const float*)d_in[30]; const float* ln3b  = (const float*)d_in[31];
    float* outp = (float*)d_out;

    // workspace carve-out
    char* p = (char*)d_ws;
    auto carve = [&](size_t bytes) -> void* {
        void* r = (void*)p;
        p += (bytes + 255) & ~(size_t)255;
        return r;
    };
    const size_t tokf = (size_t)NBQ * D_MODEL * sizeof(float);
    float*  Qb    = (float*)carve(tokf);
    float*  Kbuf  = (float*)carve(tokf);
    float*  Vbuf  = (float*)carve(tokf);
    float*  SA    = (float*)carve(tokf);
    float*  AO    = (float*)carve(tokf);
    float*  tgt1  = (float*)carve(tokf);
    float*  offb  = (float*)carve((size_t)NBQ * 192 * sizeof(float));
    float*  logb  = (float*)carve((size_t)NBQ * 96 * sizeof(float));
    float*  samp  = (float*)carve(tokf);
    float*  ca    = (float*)carve(tokf);
    float*  tgt2  = (float*)carve(tokf);
    float*  hid   = (float*)carve((size_t)NBQ * DFFN * sizeof(float));
    float*  ffo   = (float*)carve(tokf);
    half_t* valh  = (half_t*)carve((size_t)NMV * D_MODEL * sizeof(half_t));

    const dim3 blk(256);
    auto gtiles = [](int M, int N) { return dim3(M / 128, (N + 63) / 64); };

    // --- self-attention ---
    gemm_wmma<true ,false,false><<<gtiles(NBQ, D_MODEL), blk, 0, stream>>>(tgt, qpe, Wq, bq_, Qb,   NBQ, D_MODEL, D_MODEL);
    gemm_wmma<true ,false,false><<<gtiles(NBQ, D_MODEL), blk, 0, stream>>>(tgt, qpe, Wk, bk_, Kbuf, NBQ, D_MODEL, D_MODEL);
    gemm_wmma<false,false,false><<<gtiles(NBQ, D_MODEL), blk, 0, stream>>>(tgt, nullptr, Wv, bv_, Vbuf, NBQ, D_MODEL, D_MODEL);
    attn_self<<<BSZ * N_HEADS, blk, 0, stream>>>(Qb, Kbuf, Vbuf, SA);
    gemm_wmma<false,false,false><<<gtiles(NBQ, D_MODEL), blk, 0, stream>>>(SA, nullptr, Wo, bo_, AO, NBQ, D_MODEL, D_MODEL);
    ln_res<<<NBQ / 8, blk, 0, stream>>>(tgt, AO, ln1g, ln1b, tgt1, NBQ);

    // --- deformable cross-attention ---
    gemm_wmma<false,false,true ><<<gtiles(NMV, D_MODEL), blk, 0, stream>>>(mem, nullptr, Wvp, bvp, (void*)valh, NMV, D_MODEL, D_MODEL);
    gemm_wmma<true ,false,false><<<gtiles(NBQ, 192),     blk, 0, stream>>>(tgt1, qpe, Woff,  boff,  offb, NBQ, 192, D_MODEL);
    gemm_wmma<true ,false,false><<<gtiles(NBQ, 96),      blk, 0, stream>>>(tgt1, qpe, Wattn, battn, logb, NBQ, 96,  D_MODEL);
    deform_sample<<<NBQ, blk, 0, stream>>>(valh, refp, offb, logb, samp);
    gemm_wmma<false,false,false><<<gtiles(NBQ, D_MODEL), blk, 0, stream>>>(samp, nullptr, Wout, bout, ca, NBQ, D_MODEL, D_MODEL);
    ln_res<<<NBQ / 8, blk, 0, stream>>>(tgt1, ca, ln2g, ln2b, tgt2, NBQ);

    // --- FFN ---
    gemm_wmma<false,true ,false><<<gtiles(NBQ, DFFN),    blk, 0, stream>>>(tgt2, nullptr, W1, b1_, hid, NBQ, DFFN, D_MODEL);
    gemm_wmma<false,false,false><<<gtiles(NBQ, D_MODEL), blk, 0, stream>>>(hid, nullptr, W2, b2_, ffo, NBQ, D_MODEL, DFFN);
    ln_res<<<NBQ / 8, blk, 0, stream>>>(tgt2, ffo, ln3g, ln3b, outp, NBQ);
}
